// NUFNO_44633300140643
// MI455X (gfx1250) — matrix operand order repaired and checked by reading, not compiled, not run on hardware
//
#include <hip/hip_runtime.h>
#include <hip/hip_bf16.h>
#include <math.h>

// ---------------- problem constants ----------------
#define Lc   4
#define Cc   256
#define Mc   65
#define Mp   96      // M padded to multiple of 32 for WMMA K
#define KKc  32
#define CTc  256
#define Bc   16
#define Nc   4096
#define CMc  1024
#define GRP  4

typedef __bf16 bf16;
typedef __attribute__((ext_vector_type(16))) __bf16 v16bf;
typedef __attribute__((ext_vector_type(8)))  float  v8f;
typedef __attribute__((ext_vector_type(4)))  float  f32x4;

union frag_u { v16bf v; f32x4 q[2]; };
union pack_u { bf16 h[8]; f32x4 q; };

__device__ __forceinline__ float siluf(float x){ return x / (1.0f + __expf(-x)); }
__device__ __forceinline__ float geluf(float x){ return 0.5f*x*(1.0f + erff(x*0.70710678118654752f)); }

// A operand: 16x32 bf16 row-major, ld elements.
// lane<16: rows 0..15, K 0..7 & 16..23 ; lane>=16: same rows, K 8..15 & 24..31
__device__ __forceinline__ v16bf load_frag_a(const bf16* base, int ld, int lane){
  int row  = lane & 15;
  int koff = (lane >> 4) << 3;
  const bf16* p = base + row*ld + koff;
  frag_u u;
  u.q[0] = *(const f32x4*)(p);
  u.q[1] = *(const f32x4*)(p + 16);
  return u.v;
}

// B operand: 32x16 bf16, source stored transposed (T[col][k] row-major, ld elements).
// lane<16: col=lane, K 0..7 & 16..23 ; lane>=16: col=lane-16, K 8..15 & 24..31
__device__ __forceinline__ v16bf load_frag_b(const bf16* baseT, int ld, int lane){
  int col = lane & 15;
  int kb  = (lane >> 4) << 3;
  const bf16* p = baseT + col*ld + kb;
  frag_u u;
  u.q[0] = *(const f32x4*)(p);
  u.q[1] = *(const f32x4*)(p + 16);
  return u.v;
}

__device__ __forceinline__ v8f wmma_bf16(v16bf a, v16bf b, v8f c){
  return __builtin_amdgcn_wmma_f32_16x16x32_bf16(false, a, false, b, (short)0, c, false, false);
}

// ---------------- small kernels ----------------
__global__ void k_cvt_bf16(const float* __restrict__ s, bf16* __restrict__ d, int n){
  int i = blockIdx.x*blockDim.x + threadIdx.x;
  if (i < n) d[i] = (bf16)s[i];
}

// temb = dense1(silu(dense0(timestep_embedding(t)))) ; stemb = silu(temb)
__global__ void k_temb(const float* __restrict__ t,
                       const float* __restrict__ d0w, const float* __restrict__ d0b,
                       const float* __restrict__ d1w, const float* __restrict__ d1b,
                       float* __restrict__ temb, float* __restrict__ stemb){
  __shared__ float s0[CTc];
  __shared__ float s1[Cc];
  int b = blockIdx.x, tid = threadIdx.x;
  float tv = t[b];
  {
    const int hh = CTc/2;
    int j = tid & (hh-1);
    float freq = __expf(-logf(10000.0f) * (float)j / (float)(hh-1));
    float arg  = tv * freq;
    s0[tid] = (tid < hh) ? sinf(arg) : cosf(arg);
  }
  __syncthreads();
  float a = d0b[tid];
  const float* w0 = d0w + tid*CTc;
  for (int k=0;k<CTc;k++) a += w0[k]*s0[k];
  s1[tid] = siluf(a);
  __syncthreads();
  float a2 = d1b[tid];
  const float* w1 = d1w + tid*Cc;
  for (int k=0;k<Cc;k++) a2 += w1[k]*s1[k];
  temb [b*Cc+tid] = a2;
  stemb[b*Cc+tid] = siluf(a2);
}

// trig tables, both layouts, zero-padded for m>=65 so padded K contributes nothing
__global__ void k_trig(const float* __restrict__ x, const float* __restrict__ z_logit,
                       bf16* __restrict__ cosT, bf16* __restrict__ sinT,
                       bf16* __restrict__ cosN, bf16* __restrict__ sinN){
  int bn = blockIdx.x, m = threadIdx.x;
  int b = bn / Nc, n = bn % Nc;
  float zl = z_logit[0];
  float s  = 0.05f + 1.45f / (1.0f + __expf(-zl));
  float z  = x[(b*2+1)*Nc + n] * s;
  float cv = 0.0f, sv = 0.0f;
  if (m < Mc){ float ph = z * (float)(m - KKc); sincosf(ph, &sv, &cv); }
  cosN[((long)b*Nc + n)*Mp + m] = (bf16)cv;
  sinN[((long)b*Nc + n)*Mp + m] = (bf16)sv;
  cosT[((long)b*Mp + m)*Nc + n] = (bf16)cv;
  sinT[((long)b*Mp + m)*Nc + n] = (bf16)sv;
}

__global__ void k_lift(const float* __restrict__ x, const float* __restrict__ lw,
                       const float* __restrict__ lb, const float* __restrict__ temb,
                       float* __restrict__ h32){
  int idx = blockIdx.x*256 + threadIdx.x;
  int n = idx & (Nc-1);
  int bc = idx >> 12;
  int c = bc & (Cc-1);
  int b = bc >> 8;
  float x0 = x[(b*2+0)*Nc + n], x1 = x[(b*2+1)*Nc + n];
  h32[idx] = lw[c*2]*x0 + lw[c*2+1]*x1 + lb[c] + temb[b*Cc+c];
}

// per-layer: shift(B,C) = stemb@spec_tw^T+tb ; tshift(B,CM) = stemb@mlp_tw^T+tb
__global__ void k_shift(const float* __restrict__ stemb,
                        const float* __restrict__ stw, const float* __restrict__ stb,
                        const float* __restrict__ mtw, const float* __restrict__ mtb,
                        float* __restrict__ shiftv, float* __restrict__ tshift){
  int idx = blockIdx.x*blockDim.x + threadIdx.x;
  int b = idx / (Cc+CMc), o = idx % (Cc+CMc);
  const float* sb = stemb + b*Cc;
  if (o < Cc){
    float a = stb[o]; const float* w = stw + o*CTc;
    for (int k=0;k<CTc;k++) a += w[k]*sb[k];
    shiftv[b*Cc+o] = a;
  } else {
    int cm = o - Cc;
    float a = mtb[cm]; const float* w = mtw + cm*Cc;
    for (int k=0;k<Cc;k++) a += w[k]*sb[k];
    tshift[b*CMc+cm] = a;
  }
}

// groupnorm + silu + shift -> xf (bf16, (B,C,N) row-major: A operand of fwd transform)
__global__ void k_gn(const float* __restrict__ h32, const float* __restrict__ g,
                     const float* __restrict__ bb, const float* __restrict__ shiftv,
                     bf16* __restrict__ xf16){
  __shared__ float rs[256], rq[256];
  int bg = blockIdx.x, b = bg / GRP, gg = bg % GRP;
  const int CG = Cc / GRP;                 // 64
  const int TOT = CG * Nc;                 // 262144
  long base = ((long)b*Cc + gg*CG) * Nc;
  float sum = 0.f, sq = 0.f;
  for (int i = threadIdx.x; i < TOT; i += blockDim.x){
    float v = h32[base+i]; sum += v; sq += v*v;
  }
  rs[threadIdx.x] = sum; rq[threadIdx.x] = sq;
  __syncthreads();
  for (int off = 128; off > 0; off >>= 1){
    if (threadIdx.x < off){ rs[threadIdx.x]+=rs[threadIdx.x+off]; rq[threadIdx.x]+=rq[threadIdx.x+off]; }
    __syncthreads();
  }
  float inv = 1.0f/(float)TOT;
  float mu  = rs[0]*inv;
  float var = rq[0]*inv - mu*mu;
  float rstd = rsqrtf(var + 1e-5f);
  for (int i = threadIdx.x; i < TOT; i += blockDim.x){
    int c = gg*CG + (i >> 12);
    float v = (h32[base+i] - mu)*rstd*g[c] + bb[c];
    v = siluf(v) + shiftv[b*Cc+c];
    xf16[base+i] = (bf16)v;
  }
}

// forward NUFFT + spectral multiply: Gr/Gi (B,C,Mp) bf16 row-major (A operand of inverse)
__global__ void k_spec_fwd(const bf16* __restrict__ xf16,
                           const bf16* __restrict__ cosT, const bf16* __restrict__ sinT,
                           const float* __restrict__ wr, const float* __restrict__ wi,
                           bf16* __restrict__ Gr, bf16* __restrict__ Gi){
  int ct = blockIdx.x, b = blockIdx.y;
  int wave = threadIdx.x >> 5, lane = threadIdx.x & 31;
  int mt = wave;                                   // 0..5 (block of 192 = 6 waves)
  const bf16* A0  = xf16 + ((long)b*Cc + ct*16) * Nc;
  const bf16* Bc0 = cosT + ((long)b*Mp + mt*16) * Nc;
  const bf16* Bs0 = sinT + ((long)b*Mp + mt*16) * Nc;
  v8f accC = {}, accS = {};
  for (int k = 0; k < Nc; k += 32){
    __builtin_prefetch(Bc0 + k + 512, 0, 1);
    __builtin_prefetch(Bs0 + k + 512, 0, 1);
    v16bf a  = load_frag_a(A0  + k, Nc, lane);
    v16bf fc = load_frag_b(Bc0 + k, Nc, lane);
    v16bf fs = load_frag_b(Bs0 + k, Nc, lane);
    accC = wmma_bf16(a, fc, accC);
    accS = wmma_bf16(a, fs, accS);
  }
  int col = lane & 15, hi = lane >> 4;
  int m = mt*16 + col;
  #pragma unroll
  for (int j = 0; j < 8; j++){
    int c = ct*16 + j + 8*hi;
    float Fr =  accC[j] * (1.0f/(float)Nc);
    float Fi = -accS[j] * (1.0f/(float)Nc);
    float gr = 0.f, gi = 0.f;
    if (m < Mc){
      float a = wr[c*Mc+m], bq = wi[c*Mc+m];
      gr = Fr*a - Fi*bq;
      gi = Fr*bq + Fi*a;
    }
    Gr[((long)b*Cc + c)*Mp + m] = (bf16)gr;
    Gi[((long)b*Cc + c)*Mp + m] = (bf16)gi;
  }
}

// inverse NUFFT + bias + skip: h = y + bias + skip*(xf - shift); also hT16 (B,N,C)
__global__ void k_spec_inv(const bf16* __restrict__ Gr, const bf16* __restrict__ Gi,
                           const bf16* __restrict__ cosN, const bf16* __restrict__ sinN,
                           const bf16* __restrict__ xf16, const float* __restrict__ shiftv,
                           const float* __restrict__ bias, const float* __restrict__ skw,
                           float* __restrict__ h32, bf16* __restrict__ hT16){
  int ct = blockIdx.x, b = blockIdx.y;
  int wave = threadIdx.x >> 5, lane = threadIdx.x & 31;
  const bf16* Ar = Gr + ((long)b*Cc + ct*16) * Mp;
  const bf16* Ai = Gi + ((long)b*Cc + ct*16) * Mp;
  v16bf ar[3], ai[3];
  #pragma unroll
  for (int kk = 0; kk < 3; kk++){
    ar[kk] = load_frag_a(Ar + kk*32, Mp, lane);
    ai[kk] = load_frag_a(Ai + kk*32, Mp, lane);
  }
  int col = lane & 15, hi = lane >> 4;
  for (int nt = wave; nt < Nc/16; nt += 8){
    const bf16* Bc0 = cosN + ((long)b*Nc + nt*16) * Mp;
    const bf16* Bs0 = sinN + ((long)b*Nc + nt*16) * Mp;
    v8f aC = {}, aS = {};
    #pragma unroll
    for (int kk = 0; kk < 3; kk++){
      aC = wmma_bf16(ar[kk], load_frag_b(Bc0 + kk*32, Mp, lane), aC);
      aS = wmma_bf16(ai[kk], load_frag_b(Bs0 + kk*32, Mp, lane), aS);
    }
    int n = nt*16 + col;
    pack_u u;
    #pragma unroll
    for (int j = 0; j < 8; j++){
      int c = ct*16 + j + 8*hi;
      float hn = (float)xf16[((long)b*Cc + c)*Nc + n] - shiftv[b*Cc+c];
      float v  = (aC[j] - aS[j]) * (1.0f/(float)Mc) + bias[c] + skw[c]*hn;
      h32[((long)b*Cc + c)*Nc + n] = v;
      u.h[j] = (bf16)v;
    }
    *(f32x4*)&hT16[((long)b*Nc + n)*Cc + ct*16 + 8*hi] = u.q;
  }
}

// MLP layer 1: hmT(B,N,CM) = gelu(W1 h + b1 + tshift)
__global__ void k_mlp1(const bf16* __restrict__ W1bf, const bf16* __restrict__ hT16,
                       const float* __restrict__ b1, const float* __restrict__ tshift,
                       bf16* __restrict__ hmT){
  int cmt = blockIdx.x, b = blockIdx.y;
  int wave = threadIdx.x >> 5, lane = threadIdx.x & 31;
  const bf16* A0 = W1bf + (long)cmt*16*Cc;
  v16bf af[8];
  #pragma unroll
  for (int k = 0; k < 8; k++) af[k] = load_frag_a(A0 + k*32, Cc, lane);
  int col = lane & 15, hi = lane >> 4;
  for (int nt = wave; nt < Nc/16; nt += 8){
    const bf16* B0 = hT16 + ((long)b*Nc + nt*16) * Cc;
    __builtin_prefetch(B0 + 8*16*Cc, 0, 1);
    v8f acc = {};
    #pragma unroll
    for (int k = 0; k < 8; k++)
      acc = wmma_bf16(af[k], load_frag_b(B0 + k*32, Cc, lane), acc);
    int n = nt*16 + col;
    pack_u u;
    #pragma unroll
    for (int j = 0; j < 8; j++){
      int cm = cmt*16 + j + 8*hi;
      u.h[j] = (bf16)geluf(acc[j] + b1[cm] + tshift[b*CMc + cm]);
    }
    *(f32x4*)&hmT[((long)b*Nc + n)*CMc + cmt*16 + 8*hi] = u.q;
  }
}

// MLP layer 2 + residual: h += W2 hm + b2 ; writes h32 (B,C,N) and hT16 (B,N,C)
__global__ void k_mlp2(const bf16* __restrict__ W2bf, const bf16* __restrict__ hmT,
                       const float* __restrict__ b2,
                       float* __restrict__ h32, bf16* __restrict__ hT16){
  __shared__ bf16 sA[16*CMc];                       // 32 KiB weight tile, reused by all waves
  int ct = blockIdx.x, b = blockIdx.y;
  const bf16* W0 = W2bf + (long)ct*16*CMc;
  for (int v = threadIdx.x; v < (16*CMc)/8; v += blockDim.x)
    ((f32x4*)sA)[v] = ((const f32x4*)W0)[v];
  __syncthreads();
  int wave = threadIdx.x >> 5, lane = threadIdx.x & 31;
  int col = lane & 15, hi = lane >> 4;
  for (int nt = wave; nt < Nc/16; nt += 8){
    const bf16* B0 = hmT + ((long)b*Nc + nt*16) * CMc;
    v8f acc = {};
    for (int k = 0; k < CMc; k += 32){
      v16bf a = load_frag_a(sA + k, CMc, lane);     // LDS -> ds_load_b128
      acc = wmma_bf16(a, load_frag_b(B0 + k, CMc, lane), acc);
    }
    int n = nt*16 + col;
    pack_u u;
    #pragma unroll
    for (int j = 0; j < 8; j++){
      int c = ct*16 + j + 8*hi;
      long hidx = ((long)b*Cc + c)*Nc + n;
      float v = h32[hidx] + acc[j] + b2[c];
      h32[hidx] = v;
      u.h[j] = (bf16)v;
    }
    *(f32x4*)&hT16[((long)b*Nc + n)*Cc + ct*16 + 8*hi] = u.q;
  }
}

// projection: out[b,n] = proj_w2 . gelu(proj_w1 h + b1) + b2
__global__ void k_proj(const bf16* __restrict__ P1bf, const bf16* __restrict__ hT16,
                       const float* __restrict__ pb1, const float* __restrict__ pw2,
                       const float* __restrict__ pb2, float* __restrict__ out){
  int b = blockIdx.y;
  int wave = threadIdx.x >> 5, lane = threadIdx.x & 31;
  int nt = blockIdx.x*8 + wave;
  const bf16* B0 = hT16 + ((long)b*Nc + nt*16) * Cc;
  v16bf bfr[8];
  #pragma unroll
  for (int k = 0; k < 8; k++) bfr[k] = load_frag_b(B0 + k*32, Cc, lane);
  int col = lane & 15, hi = lane >> 4;
  float part = 0.f;
  for (int ctt = 0; ctt < 16; ctt++){
    const bf16* A0 = P1bf + (long)ctt*16*Cc;
    v8f acc = {};
    #pragma unroll
    for (int k = 0; k < 8; k++)
      acc = wmma_bf16(load_frag_a(A0 + k*32, Cc, lane), bfr[k], acc);
    #pragma unroll
    for (int j = 0; j < 8; j++){
      int cto = ctt*16 + j + 8*hi;
      part += pw2[cto] * geluf(acc[j] + pb1[cto]);
    }
  }
  part += __shfl_xor(part, 16, 32);                 // combine row halves
  if (hi == 0) out[b*Nc + nt*16 + col] = part + pb2[0];
}

// ---------------- host launch ----------------
extern "C" void kernel_launch(void* const* d_in, const int* in_sizes, int n_in,
                              void* d_out, int out_size, void* d_ws, size_t ws_size,
                              hipStream_t stream){
  (void)in_sizes; (void)n_in; (void)out_size; (void)ws_size;
  const float* x       = (const float*)d_in[0];
  const float* t       = (const float*)d_in[1];
  const float* lift_w  = (const float*)d_in[2];
  const float* lift_b  = (const float*)d_in[3];
  const float* d0w     = (const float*)d_in[4];
  const float* d0b     = (const float*)d_in[5];
  const float* d1w     = (const float*)d_in[6];
  const float* d1b     = (const float*)d_in[7];
  const float* spec_wr = (const float*)d_in[8];
  const float* spec_wi = (const float*)d_in[9];
  const float* spec_bias=(const float*)d_in[10];
  const float* spec_tw = (const float*)d_in[11];
  const float* spec_tb = (const float*)d_in[12];
  const float* norm_g  = (const float*)d_in[13];
  const float* norm_bb = (const float*)d_in[14];
  const float* skip_w  = (const float*)d_in[15];
  const float* mlp_w1  = (const float*)d_in[16];
  const float* mlp_b1  = (const float*)d_in[17];
  const float* mlp_tw  = (const float*)d_in[18];
  const float* mlp_tb  = (const float*)d_in[19];
  const float* mlp_w2  = (const float*)d_in[20];
  const float* mlp_b2  = (const float*)d_in[21];
  const float* proj_w1 = (const float*)d_in[22];
  const float* proj_b1 = (const float*)d_in[23];
  const float* proj_w2 = (const float*)d_in[24];
  const float* proj_b2 = (const float*)d_in[25];
  const float* z_logit = (const float*)d_in[26];

  char* wp = (char*)d_ws;
  auto carve = [&](size_t bytes)->char*{
    char* p = wp; wp += (bytes + 255) & ~(size_t)255; return p;
  };
  float* temb   = (float*)carve((size_t)Bc*Cc*4);
  float* stemb  = (float*)carve((size_t)Bc*Cc*4);
  float* shiftv = (float*)carve((size_t)Bc*Cc*4);
  float* tshift = (float*)carve((size_t)Bc*CMc*4);
  float* h32    = (float*)carve((size_t)Bc*Cc*Nc*4);      // 64 MiB
  bf16*  xf16   = (bf16*) carve((size_t)Bc*Cc*Nc*2);      // 32 MiB
  bf16*  hT16   = (bf16*) carve((size_t)Bc*Nc*Cc*2);      // 32 MiB
  bf16*  hmT    = (bf16*) carve((size_t)Bc*Nc*CMc*2);     // 128 MiB
  bf16*  cosT   = (bf16*) carve((size_t)Bc*Mp*Nc*2);
  bf16*  sinT   = (bf16*) carve((size_t)Bc*Mp*Nc*2);
  bf16*  cosN   = (bf16*) carve((size_t)Bc*Nc*Mp*2);
  bf16*  sinN   = (bf16*) carve((size_t)Bc*Nc*Mp*2);
  bf16*  Grb    = (bf16*) carve((size_t)Bc*Cc*Mp*2);
  bf16*  Gib    = (bf16*) carve((size_t)Bc*Cc*Mp*2);
  bf16*  W1bf   = (bf16*) carve((size_t)Lc*CMc*Cc*2);
  bf16*  W2bf   = (bf16*) carve((size_t)Lc*Cc*CMc*2);
  bf16*  P1bf   = (bf16*) carve((size_t)CTc*Cc*2);

  { int n = Lc*CMc*Cc; k_cvt_bf16<<<(n+255)/256,256,0,stream>>>(mlp_w1, W1bf, n); }
  { int n = Lc*Cc*CMc; k_cvt_bf16<<<(n+255)/256,256,0,stream>>>(mlp_w2, W2bf, n); }
  { int n = CTc*Cc;    k_cvt_bf16<<<(n+255)/256,256,0,stream>>>(proj_w1, P1bf, n); }

  k_temb<<<Bc, 256, 0, stream>>>(t, d0w, d0b, d1w, d1b, temb, stemb);
  k_trig<<<Bc*Nc, Mp, 0, stream>>>(x, z_logit, cosT, sinT, cosN, sinN);
  k_lift<<<(Bc*Cc*Nc)/256, 256, 0, stream>>>(x, lift_w, lift_b, temb, h32);

  for (int i = 0; i < Lc; i++){
    k_shift<<<(Bc*(Cc+CMc))/256, 256, 0, stream>>>(
        stemb, spec_tw + (size_t)i*Cc*CTc, spec_tb + i*Cc,
        mlp_tw + (size_t)i*CMc*Cc, mlp_tb + i*CMc, shiftv, tshift);
    k_gn<<<Bc*GRP, 256, 0, stream>>>(h32, norm_g + i*Cc, norm_bb + i*Cc, shiftv, xf16);
    k_spec_fwd<<<dim3(Cc/16, Bc), 192, 0, stream>>>(
        xf16, cosT, sinT, spec_wr + (size_t)i*Cc*Mc, spec_wi + (size_t)i*Cc*Mc, Grb, Gib);
    k_spec_inv<<<dim3(Cc/16, Bc), 256, 0, stream>>>(
        Grb, Gib, cosN, sinN, xf16, shiftv, spec_bias + i*Cc, skip_w + i*Cc, h32, hT16);
    k_mlp1<<<dim3(CMc/16, Bc), 256, 0, stream>>>(
        W1bf + (size_t)i*CMc*Cc, hT16, mlp_b1 + i*CMc, tshift, hmT);
    k_mlp2<<<dim3(Cc/16, Bc), 256, 0, stream>>>(
        W2bf + (size_t)i*Cc*CMc, hmT, mlp_b2 + i*Cc, h32, hT16);
  }

  k_proj<<<dim3(Nc/16/8, Bc), 256, 0, stream>>>(
      P1bf, hT16, proj_b1, proj_w2, proj_b2, (float*)d_out);
}
